// CKANCore_74921409511942
// MI455X (gfx1250) — compile-verified, matrix operand order checked by reading, not run on hardware
//
#include <hip/hip_runtime.h>
#include <hip/hip_bf16.h>
#include <math.h>

typedef _Float16 v16h __attribute__((ext_vector_type(16)));
typedef _Float16 v8h  __attribute__((ext_vector_type(8)));
typedef float    v8f  __attribute__((ext_vector_type(8)));

#define DIM     64
#define DIM2    128
#define N_MEM   64
#define BATCH   4096
#define NREL    32

// Shared memory layout (aligned for 16-byte v8h LDS loads).
struct __align__(16) SMem {
    _Float16 X[N_MEM * DIM2];    // 16 KB: input rows [h|r] f16; reused as f32 A2[64][64]
    _Float16 W1T[DIM * DIM2];    // 16 KB: W1 transposed (row c = column c of W1)
    _Float16 W2T[DIM * DIM];     //  8 KB
    _Float16 A1[N_MEM * DIM];    //  8 KB: ReLU(X @ W1) in f16
    float    W3[DIM];
    float    score[N_MEM];
    float    aw[N_MEM];
    int      hidx[N_MEM];
    int      ridx[N_MEM];
    int      tidx[N_MEM];
};

// Load a 16-bit WMMA A/B fragment (16x32 tile slice for one lane) from an LDS
// row: elements 0..7 = K lo..lo+7, elements 8..15 = K lo+16..lo+23.
__device__ __forceinline__ v16h load_frag(const _Float16* row, int lo) {
    v16h f;
    *(v8h*)&f       = *(const v8h*)(row + lo);
    *((v8h*)&f + 1) = *(const v8h*)(row + lo + 16);
    return f;
}

__device__ __forceinline__ v8f wmma16(v16h a, v16h b, v8f c) {
    return __builtin_amdgcn_wmma_f32_16x16x32_f16(false, a, false, b,
                                                  (short)0, c, false, false);
}

__global__ void __launch_bounds__(128)
ckan_attention_kernel(const int* __restrict__ user_h, const int* __restrict__ user_r,
                      const int* __restrict__ user_t, const int* __restrict__ item_h,
                      const int* __restrict__ item_r, const int* __restrict__ item_t,
                      const float* __restrict__ entity, const float* __restrict__ relation,
                      const float* __restrict__ W1, const float* __restrict__ W2,
                      const float* __restrict__ W3g, float* __restrict__ ws)
{
    __shared__ SMem sm;
    const int tid   = threadIdx.x;
    const int wg    = blockIdx.x;          // 0..16383
    const int b     = wg & (BATCH - 1);
    const int sl    = wg >> 12;            // side*2 + layer
    const int side  = sl >> 1;             // 0 = user, 1 = item
    const int layer = sl & 1;

    const int* hI = side ? item_h : user_h;
    const int* rI = side ? item_r : user_r;
    const int* tI = side ? item_t : user_t;
    const int  base = (layer * BATCH + b) * N_MEM;

    // ---- Stage indices -------------------------------------------------
    if (tid < N_MEM) {
        sm.hidx[tid] = hI[base + tid];
        sm.ridx[tid] = rI[base + tid];
        sm.tidx[tid] = tI[base + tid];
    }
    __syncthreads();

    // ---- Stage X = [h | r] rows (f32 -> f16), accumulate head-mean -----
    // Thread k<64 streams h[:, k] (waves 0-1); thread k>=64 streams r[:, k-64].
    float hsum = 0.f;
    for (int m = 0; m < N_MEM; ++m) {
        float v;
        if (tid < DIM) {
            v = entity[sm.hidx[m] * DIM + tid];
            hsum += v;
        } else {
            v = relation[sm.ridx[m] * DIM + (tid - DIM)];
        }
        sm.X[m * DIM2 + tid] = (_Float16)v;
    }
    const float hmean = hsum * (1.0f / (float)N_MEM);

    // ---- Stage weights (transposed, f16) -------------------------------
    for (int i = tid; i < DIM2 * DIM; i += 128) {      // W1: [128][64] row-major
        int k = i >> 6, c = i & 63;
        sm.W1T[c * DIM2 + k] = (_Float16)W1[i];
    }
    for (int i = tid; i < DIM * DIM; i += 128) {       // W2: [64][64]
        int k = i >> 6, c = i & 63;
        sm.W2T[c * DIM + k] = (_Float16)W2[i];
    }
    if (tid < DIM) sm.W3[tid] = W3g[tid];
    __syncthreads();

    // ---- GEMM1: A1 = ReLU(X[64x128] @ W1[128x64]) via WMMA -------------
    const int lane  = tid & 31;
    const int hs    = lane >> 4;        // half-select within wave
    const int mr    = lane & 15;
    const int Mbase = (tid >> 5) * 16;  // wave's 16-row slab

    v16h af[4];
    {
        const _Float16* xr = &sm.X[(Mbase + mr) * DIM2];
        #pragma unroll
        for (int kb = 0; kb < 4; ++kb)
            af[kb] = load_frag(xr, kb * 32 + hs * 8);
    }
    #pragma unroll
    for (int nt = 0; nt < 4; ++nt) {
        const _Float16* wr = &sm.W1T[(nt * 16 + mr) * DIM2];
        v8f acc = {};
        #pragma unroll
        for (int kb = 0; kb < 4; ++kb)
            acc = wmma16(af[kb], load_frag(wr, kb * 32 + hs * 8), acc);
        #pragma unroll
        for (int v = 0; v < 8; ++v) {
            int m = Mbase + v + hs * 8;
            int n = nt * 16 + mr;
            sm.A1[m * DIM + n] = (_Float16)fmaxf(acc[v], 0.f);
        }
    }
    __syncthreads();

    // ---- GEMM2: A2 = ReLU(A1[64x64] @ W2[64x64]) -----------------------
    float* A2 = (float*)sm.X;   // reuse X region (16 KB) for f32 A2
    v16h af2[2];
    {
        const _Float16* ar = &sm.A1[(Mbase + mr) * DIM];
        af2[0] = load_frag(ar, hs * 8);
        af2[1] = load_frag(ar, 32 + hs * 8);
    }
    #pragma unroll
    for (int nt = 0; nt < 4; ++nt) {
        const _Float16* wr = &sm.W2T[(nt * 16 + mr) * DIM];
        v8f acc = {};
        acc = wmma16(af2[0], load_frag(wr, hs * 8), acc);
        acc = wmma16(af2[1], load_frag(wr, 32 + hs * 8), acc);
        #pragma unroll
        for (int v = 0; v < 8; ++v) {
            int m = Mbase + v + hs * 8;
            int n = nt * 16 + mr;
            A2[m * DIM + n] = fmaxf(acc[v], 0.f);
        }
    }
    __syncthreads();

    // ---- Attention scores: sigmoid(A2 @ W3), then softmax over m -------
    if (tid < N_MEM) {
        float s = 0.f;
        #pragma unroll 8
        for (int n = 0; n < DIM; ++n) s += A2[tid * DIM + n] * sm.W3[n];
        sm.score[tid] = 1.f / (1.f + __expf(-s));
    }
    __syncthreads();
    if (tid < N_MEM) {
        float mx = sm.score[0];
        #pragma unroll 8
        for (int m = 1; m < N_MEM; ++m) mx = fmaxf(mx, sm.score[m]);
        float sum = 0.f;
        #pragma unroll 8
        for (int m = 0; m < N_MEM; ++m) sum += __expf(sm.score[m] - mx);
        sm.aw[tid] = __expf(sm.score[tid] - mx) / sum;
    }
    __syncthreads();

    // ---- Weighted t-sum (+ layer-0 head-mean), write workspace slot ----
    if (tid < DIM) {
        float acc = (layer == 0) ? hmean : 0.f;
        #pragma unroll 4
        for (int m = 0; m < N_MEM; ++m)
            acc += sm.aw[m] * entity[sm.tidx[m] * DIM + tid];
        ws[((size_t)sl * BATCH + b) * DIM + tid] = acc;
    }
}

__global__ void __launch_bounds__(64)
ckan_combine_kernel(const int* __restrict__ items, const float* __restrict__ entity,
                    const float* __restrict__ ws, float* __restrict__ out)
{
    const int b = blockIdx.x;
    const int d = threadIdx.x;
    __shared__ float red[DIM];

    // slots: 0 = user L0 (+head-mean), 1 = user L1, 2 = item L0 (+head-mean), 3 = item L1
    float eu = ws[((size_t)0 * BATCH + b) * DIM + d] + ws[((size_t)1 * BATCH + b) * DIM + d];
    float ev = entity[items[b] * DIM + d]
             + ws[((size_t)2 * BATCH + b) * DIM + d]
             + ws[((size_t)3 * BATCH + b) * DIM + d];
    red[d] = eu * ev;
    __syncthreads();
    #pragma unroll
    for (int s = 32; s > 0; s >>= 1) {
        if (d < s) red[d] += red[d + s];
        __syncthreads();
    }
    if (d == 0) out[b] = 1.f / (1.f + __expf(-red[0]));
}

extern "C" void kernel_launch(void* const* d_in, const int* in_sizes, int n_in,
                              void* d_out, int out_size, void* d_ws, size_t ws_size,
                              hipStream_t stream)
{
    const int*   items    = (const int*)d_in[0];
    const int*   user_h   = (const int*)d_in[1];
    const int*   user_r   = (const int*)d_in[2];
    const int*   user_t   = (const int*)d_in[3];
    const int*   item_h   = (const int*)d_in[4];
    const int*   item_r   = (const int*)d_in[5];
    const int*   item_t   = (const int*)d_in[6];
    const float* entity   = (const float*)d_in[7];
    const float* relation = (const float*)d_in[8];
    const float* W1       = (const float*)d_in[9];
    const float* W2       = (const float*)d_in[10];
    const float* W3       = (const float*)d_in[11];
    float* out = (float*)d_out;
    float* ws  = (float*)d_ws;   // needs 4 * 4096 * 64 * 4 B = 4 MB

    ckan_attention_kernel<<<dim3(4 * BATCH), dim3(128), 0, stream>>>(
        user_h, user_r, user_t, item_h, item_r, item_t,
        entity, relation, W1, W2, W3, ws);

    ckan_combine_kernel<<<dim3(BATCH), dim3(64), 0, stream>>>(items, entity, ws, out);
}
